// MultiscaleGraphTransformer_14396730376428
// MI455X (gfx1250) — matrix operand order repaired and checked by reading, not compiled, not run on hardware
//
#include <hip/hip_runtime.h>
#include <hip/hip_bf16.h>
#include <math.h>

typedef __attribute__((ext_vector_type(16))) _Float16 v16h;
typedef __attribute__((ext_vector_type(8)))  _Float16 v8h;
typedef __attribute__((ext_vector_type(8)))  float    v8f;

#define BATCH 16
#define SEQ   512
#define DMODEL 256
#define NHEAD 8
#define HD 32
#define MROWS (BATCH*SEQ)   // 8192

__device__ __forceinline__ float gelu_exact(float x) {
    return 0.5f * x * (1.0f + erff(x * 0.70710678118654752f));
}

__device__ __forceinline__ float pe_val(int s, int n) {
    int j = n & ~1;
    float div = expf((float)j * (-9.210340371976184f / 256.0f)); // -ln(10000)/DM
    float ang = (float)s * div;
    return (n & 1) ? cosf(ang) : sinf(ang);
}

__device__ __forceinline__ float wave_sum32(float v) {
    #pragma unroll
    for (int off = 16; off > 0; off >>= 1) v += __shfl_xor(v, off, 32);
    return v;
}

// ---------------------------------------------------------------------------
// fp32 -> fp16 weight conversion (once per call; weights are static).
// ---------------------------------------------------------------------------
__global__ void wconv_kernel(const float* __restrict__ src,
                             _Float16* __restrict__ dst, int n)
{
    for (int i = blockIdx.x * blockDim.x + threadIdx.x; i < n;
         i += gridDim.x * blockDim.x)
        dst[i] = (_Float16)src[i];
}

// ---------------------------------------------------------------------------
// Generic WMMA GEMM: C[row, colOff+n] = epi( A[row,:K] . W[n,:K] + bias[n] )
// A fp32 [M x lda]; W fp16 [N x ldw] pre-converted (row n = output feature).
// Template params keep all per-element code branch-free:
//   ADDK : add addK[k] along K to A rows (scale_emb)
//   RESID: add residual
//   FLAGS: 1 = GELU, 2 = sinusoidal positional encoding (row % SEQ)
// One wave computes a 16x64 tile: one A-frag (vectorized float4 loads),
// four B-frags in distinct registers, four back-to-back WMMAs.
// ---------------------------------------------------------------------------
template<bool ADDK, bool RESID, int FLAGS>
__global__ void gemm_wmma_kernel(
    const float* __restrict__ A, int lda, const float* __restrict__ addK,
    const _Float16* __restrict__ W, int ldw,
    const float* __restrict__ bias,
    const float* __restrict__ resid, int ldr,
    float* __restrict__ C, int ldc, int colOff,
    int N, int K)
{
    const int wid  = blockIdx.x * (blockDim.x >> 5) + (threadIdx.x >> 5);
    const int lane = threadIdx.x & 31;
    const int lh   = lane >> 4;     // lane-half select
    const int ln   = lane & 15;
    const int ngrp = N >> 6;        // N / 64
    const int mt   = wid / ngrp;
    const int ng   = wid % ngrp;

    const float* arow = A + (size_t)(mt * 16 + ln) * lda + 16 * lh;
    const float* krow = ADDK ? (addK + 16 * lh) : nullptr;
    const _Float16* wbase0 = W + (size_t)(ng * 64 + ln) * ldw + 16 * lh;
    const _Float16* wbase1 = wbase0 + (size_t)16 * ldw;
    const _Float16* wbase2 = wbase0 + (size_t)32 * ldw;
    const _Float16* wbase3 = wbase0 + (size_t)48 * ldw;

    v8f c0 = {}, c1 = {}, c2 = {}, c3 = {};

    for (int k0 = 0; k0 < K; k0 += 32) {
        // A fragment: one contiguous 16-float run at k0+16*lh.
        float4 a0 = *(const float4*)(arow + k0);
        float4 a1 = *(const float4*)(arow + k0 + 4);
        float4 a2 = *(const float4*)(arow + k0 + 8);
        float4 a3 = *(const float4*)(arow + k0 + 12);
        if (ADDK) {
            float4 s0 = *(const float4*)(krow + k0);
            float4 s1 = *(const float4*)(krow + k0 + 4);
            float4 s2 = *(const float4*)(krow + k0 + 8);
            float4 s3 = *(const float4*)(krow + k0 + 12);
            a0.x += s0.x; a0.y += s0.y; a0.z += s0.z; a0.w += s0.w;
            a1.x += s1.x; a1.y += s1.y; a1.z += s1.z; a1.w += s1.w;
            a2.x += s2.x; a2.y += s2.y; a2.z += s2.z; a2.w += s2.w;
            a3.x += s3.x; a3.y += s3.y; a3.z += s3.z; a3.w += s3.w;
        }
        v16h a;
        a[0]  = (_Float16)a0.x; a[1]  = (_Float16)a0.y;
        a[2]  = (_Float16)a0.z; a[3]  = (_Float16)a0.w;
        a[4]  = (_Float16)a1.x; a[5]  = (_Float16)a1.y;
        a[6]  = (_Float16)a1.z; a[7]  = (_Float16)a1.w;
        a[8]  = (_Float16)a2.x; a[9]  = (_Float16)a2.y;
        a[10] = (_Float16)a2.z; a[11] = (_Float16)a2.w;
        a[12] = (_Float16)a3.x; a[13] = (_Float16)a3.y;
        a[14] = (_Float16)a3.z; a[15] = (_Float16)a3.w;

        // Four B fragments in distinct registers (32B-aligned vector loads).
        v16h b0 = *(const v16h*)(wbase0 + k0);
        v16h b1 = *(const v16h*)(wbase1 + k0);
        v16h b2 = *(const v16h*)(wbase2 + k0);
        v16h b3 = *(const v16h*)(wbase3 + k0);
        c0 = __builtin_amdgcn_wmma_f32_16x16x32_f16(false, a, false, b0, (short)0, c0, false, false);
        c1 = __builtin_amdgcn_wmma_f32_16x16x32_f16(false, a, false, b1, (short)0, c1, false, false);
        c2 = __builtin_amdgcn_wmma_f32_16x16x32_f16(false, a, false, b2, (short)0, c2, false, false);
        c3 = __builtin_amdgcn_wmma_f32_16x16x32_f16(false, a, false, b3, (short)0, c3, false, false);
    }

    v8f cacc[4] = {c0, c1, c2, c3};
    #pragma unroll
    for (int t = 0; t < 4; t++) {
        const int n = ng * 64 + t * 16 + ln;
        const float bv = bias[n];
        #pragma unroll
        for (int j = 0; j < 8; j++) {
            const int row = mt * 16 + j + 8 * lh;
            float v = cacc[t][j] + bv;
            if (RESID) v += resid[(size_t)row * ldr + n];
            if (FLAGS == 1) v = gelu_exact(v);
            if (FLAGS == 2) v += pe_val(row % SEQ, n);
            C[(size_t)row * ldc + colOff + n] = v;
        }
    }
}

// ---------------------------------------------------------------------------
// Banded attention: one wave per (b, h, 16-query tile). 128-key window covers
// the widest band (16 + 2*50 = 116). Scores computed transposed (K-tile x Q)
// so softmax is a per-lane LDS column scan; P.V accumulated with WMMA.
// Loops capped at unroll-2 to avoid register spills (unroll-8 spilled).
// qkv layout: [B*S][768] = [q(256) | k(256) | v(256)], head h at h*32.
// ---------------------------------------------------------------------------
__global__ void attn_kernel(const float* __restrict__ qkv,
                            float* __restrict__ o, int sc)
{
    __shared__ float    sS[128 * 16];   // scores [key_local][q]
    __shared__ _Float16 sP[16 * 128];   // probs  [q][key_local] (unnormalized)
    __shared__ float    sInv[16];       // 1/rowsum

    const int wid = blockIdx.x;
    const int qt = wid & 31;
    const int h  = (wid >> 5) & 7;
    const int b  = wid >> 8;
    const int lane = threadIdx.x & 31;
    const int lh = lane >> 4, ln = lane & 15;
    const int q0 = qt * 16;
    int kstart = q0 - 56;
    if (kstart < 0) kstart = 0;
    if (kstart > SEQ - 128) kstart = SEQ - 128;

    const float* base = qkv + (size_t)(b * SEQ) * 768 + h * 32;

    auto cvt16 = [](float4 x0, float4 x1, float4 x2, float4 x3) -> v16h {
        v16h r;
        r[0]=(_Float16)x0.x; r[1]=(_Float16)x0.y; r[2]=(_Float16)x0.z; r[3]=(_Float16)x0.w;
        r[4]=(_Float16)x1.x; r[5]=(_Float16)x1.y; r[6]=(_Float16)x1.z; r[7]=(_Float16)x1.w;
        r[8]=(_Float16)x2.x; r[9]=(_Float16)x2.y; r[10]=(_Float16)x2.z; r[11]=(_Float16)x2.w;
        r[12]=(_Float16)x3.x; r[13]=(_Float16)x3.y; r[14]=(_Float16)x3.z; r[15]=(_Float16)x3.w;
        return r;
    };

    // B fragment = Q^T (hd is the WMMA K-dim, exactly 32)
    v16h bq;
    {
        const float* qrow = base + (size_t)(q0 + ln) * 768 + 8 * lh;
        bq = cvt16(*(const float4*)(qrow),     *(const float4*)(qrow + 4),
                   *(const float4*)(qrow + 16), *(const float4*)(qrow + 20));
    }

    const float scl = 0.17677669529663687f; // 1/sqrt(32)
    #pragma unroll 2
    for (int kt = 0; kt < 8; kt++) {
        // A fragment: 16 key rows x 32 hd (two contiguous 8-float runs)
        const float* krow = base + (size_t)(kstart + kt * 16 + ln) * 768 + 256 + 8 * lh;
        v16h ak = cvt16(*(const float4*)(krow),      *(const float4*)(krow + 4),
                        *(const float4*)(krow + 16), *(const float4*)(krow + 20));
        v8f c = {};
        c = __builtin_amdgcn_wmma_f32_16x16x32_f16(
            false, ak, false, bq, (short)0, c, false, false);
        #pragma unroll
        for (int j = 0; j < 8; j++)
            sS[(kt * 16 + j + 8 * lh) * 16 + ln] = c[j] * scl;
    }
    __syncthreads();

    if (lane < 16) {
        const int q = lane, qpos = q0 + q;
        float mx = -1e30f;
        for (int key = 0; key < 128; key++) {
            int d = kstart + key - qpos; if (d < 0) d = -d;
            if (d <= sc) { float v = sS[key * 16 + q]; mx = v > mx ? v : mx; }
        }
        float sum = 0.0f;
        for (int key = 0; key < 128; key++) {
            int d = kstart + key - qpos; if (d < 0) d = -d;
            float e = (d <= sc) ? __expf(sS[key * 16 + q] - mx) : 0.0f;
            sP[q * 128 + key] = (_Float16)e;
            sum += e;
        }
        sInv[q] = 1.0f / sum;
    }
    __syncthreads();

    v8f c0 = {}, c1 = {};
    #pragma unroll 2
    for (int ch = 0; ch < 4; ch++) {
        // A fragment: probs [16 q x 32 keys], one contiguous 16-half LDS run at 16*lh
        v8h p0 = *(const v8h*)(sP + ln * 128 + ch * 32 + 16 * lh);
        v8h p1 = *(const v8h*)(sP + ln * 128 + ch * 32 + 8 + 16 * lh);
        v16h pa = __builtin_shufflevector(p0, p1,
            0, 1, 2, 3, 4, 5, 6, 7, 8, 9, 10, 11, 12, 13, 14, 15);

        v16h b0, b1; // B fragments: V [32 keys x 16 hd] x2 halves
        #pragma unroll
        for (int i = 0; i < 16; i++) {
            const float* vrow = base + (size_t)(kstart + ch * 32 + 16 * lh + i) * 768 + 512;
            b0[i] = (_Float16)vrow[ln];
            b1[i] = (_Float16)vrow[16 + ln];
        }
        c0 = __builtin_amdgcn_wmma_f32_16x16x32_f16(false, pa, false, b0, (short)0, c0, false, false);
        c1 = __builtin_amdgcn_wmma_f32_16x16x32_f16(false, pa, false, b1, (short)0, c1, false, false);
    }
    #pragma unroll
    for (int j = 0; j < 8; j++) {
        const int q = j + 8 * lh;
        const float inv = sInv[q];
        float* orow = o + (size_t)(b * SEQ + q0 + q) * DMODEL + h * 32;
        orow[ln]      = c0[j] * inv;
        orow[16 + ln] = c1[j] * inv;
    }
}

// ---------------------------------------------------------------------------
// Row LayerNorm over DMODEL=256: one wave per row, 8 elems/lane.
// ---------------------------------------------------------------------------
__global__ void ln_kernel(const float* __restrict__ in, float* __restrict__ out,
                          const float* __restrict__ g, const float* __restrict__ b)
{
    const int row  = blockIdx.x * (blockDim.x >> 5) + (threadIdx.x >> 5);
    const int lane = threadIdx.x & 31;
    const float* r = in + (size_t)row * DMODEL;
    float v[8], s = 0.0f;
    #pragma unroll
    for (int i = 0; i < 8; i++) { v[i] = r[lane + i * 32]; s += v[i]; }
    const float mean = wave_sum32(s) * (1.0f / 256.0f);
    float var = 0.0f;
    #pragma unroll
    for (int i = 0; i < 8; i++) { float d = v[i] - mean; var += d * d; }
    var = wave_sum32(var) * (1.0f / 256.0f);
    const float inv = rsqrtf(var + 1e-5f);
    float* w = out + (size_t)row * DMODEL;
    #pragma unroll
    for (int i = 0; i < 8; i++) {
        const int n = lane + i * 32;
        w[n] = (v[i] - mean) * inv * g[n] + b[n];
    }
}

// ---------------------------------------------------------------------------
// Head: 16 rows only — single scalar workgroup.
// ---------------------------------------------------------------------------
__global__ void head_kernel(const float* __restrict__ x, const float* __restrict__ feats,
    const float* fw, const float* fb, const float* flg, const float* flb,
    const float* h1w, const float* h1b, const float* hg, const float* hb,
    const float* h2w, const float* h2b, const float* h3w, const float* h3b,
    float* __restrict__ out)
{
    __shared__ float xc[16][512];
    __shared__ float t1[16][256];
    __shared__ float h2s[16][128];
    const int tid = threadIdx.x; // 256 threads

    for (int idx = tid; idx < 16 * 256; idx += 256) {
        const int b = idx >> 8, n = idx & 255;
        xc[b][n] = x[((size_t)b * SEQ + SEQ / 2) * DMODEL + n];
    }
    for (int idx = tid; idx < 16 * 256; idx += 256) {
        const int b = idx >> 8, n = idx & 255;
        float acc = fb[n];
        for (int k = 0; k < 66; k++) acc += feats[b * 66 + k] * fw[n * 66 + k];
        t1[b][n] = acc;
    }
    __syncthreads();
    if (tid < 16) {
        const int b = tid;
        float m = 0; for (int n = 0; n < 256; n++) m += t1[b][n]; m *= (1.0f/256.0f);
        float v = 0; for (int n = 0; n < 256; n++) { float d = t1[b][n] - m; v += d * d; }
        const float inv = rsqrtf(v * (1.0f/256.0f) + 1e-5f);
        for (int n = 0; n < 256; n++)
            xc[b][256 + n] = gelu_exact((t1[b][n] - m) * inv * flg[n] + flb[n]);
    }
    __syncthreads();
    for (int idx = tid; idx < 16 * 256; idx += 256) {
        const int b = idx >> 8, n = idx & 255;
        float acc = h1b[n];
        for (int k = 0; k < 512; k++) acc += xc[b][k] * h1w[n * 512 + k];
        t1[b][n] = acc;
    }
    __syncthreads();
    if (tid < 16) {
        const int b = tid;
        float m = 0; for (int n = 0; n < 256; n++) m += t1[b][n]; m *= (1.0f/256.0f);
        float v = 0; for (int n = 0; n < 256; n++) { float d = t1[b][n] - m; v += d * d; }
        const float inv = rsqrtf(v * (1.0f/256.0f) + 1e-5f);
        for (int n = 0; n < 256; n++)
            t1[b][n] = gelu_exact((t1[b][n] - m) * inv * hg[n] + hb[n]);
    }
    __syncthreads();
    for (int idx = tid; idx < 16 * 128; idx += 256) {
        const int b = idx >> 7, n = idx & 127;
        float acc = h2b[n];
        for (int k = 0; k < 256; k++) acc += t1[b][k] * h2w[n * 256 + k];
        h2s[b][n] = gelu_exact(acc);
    }
    __syncthreads();
    if (tid < 16) {
        const int b = tid;
        float acc = h3b[0];
        for (int k = 0; k < 128; k++) acc += h2s[b][k] * h3w[k];
        out[b] = acc;
    }
}

// ---------------------------------------------------------------------------
extern "C" void kernel_launch(void* const* d_in, const int* in_sizes, int n_in,
                              void* d_out, int out_size, void* d_ws, size_t ws_size,
                              hipStream_t stream)
{
    (void)in_sizes; (void)n_in; (void)out_size; (void)ws_size;
    const float* emb        = (const float*)d_in[0];
    const float* features   = (const float*)d_in[1];
    const float* ip_w       = (const float*)d_in[2];
    const float* ip_b       = (const float*)d_in[3];
    const float* scale_emb  = (const float*)d_in[4];
    const float* attn_in_w  = (const float*)d_in[5];
    const float* attn_in_b  = (const float*)d_in[6];
    const float* attn_out_w = (const float*)d_in[7];
    const float* attn_out_b = (const float*)d_in[8];
    const float* fus_w      = (const float*)d_in[9];
    const float* fus_b      = (const float*)d_in[10];
    const float* ffn_w1     = (const float*)d_in[11];
    const float* ffn_b1     = (const float*)d_in[12];
    const float* ffn_w2     = (const float*)d_in[13];
    const float* ffn_b2     = (const float*)d_in[14];
    const float* n1_g       = (const float*)d_in[15];
    const float* n1_b       = (const float*)d_in[16];
    const float* n2_g       = (const float*)d_in[17];
    const float* n2_b       = (const float*)d_in[18];
    const float* feat_w     = (const float*)d_in[19];
    const float* feat_b     = (const float*)d_in[20];
    const float* feat_ln_g  = (const float*)d_in[21];
    const float* feat_ln_b  = (const float*)d_in[22];
    const float* h1_w       = (const float*)d_in[23];
    const float* h1_b       = (const float*)d_in[24];
    const float* h_ln_g     = (const float*)d_in[25];
    const float* h_ln_b     = (const float*)d_in[26];
    const float* h2_w       = (const float*)d_in[27];
    const float* h2_b       = (const float*)d_in[28];
    const float* h3_w       = (const float*)d_in[29];
    const float* h3_b       = (const float*)d_in[30];

    float* ws    = (float*)d_ws;
    float* r_x   = ws;                       // 8192*256
    float* r_tmp = ws + 2u * 1024 * 1024;    // 8192*256
    float* r_o   = ws + 4u * 1024 * 1024;    // 8192*256
    float* r_qkv = ws + 6u * 1024 * 1024;    // 8192*768
    float* r_cc  = ws + 12u * 1024 * 1024;   // 8192*1024 (also FFN hidden)

    // fp16 weight cache (converted each call; weights static within a call)
    _Float16* hmem = (_Float16*)(ws + 20u * 1024 * 1024);
    const int n_ip   = 256 * 640;
    const int n_ain  = 16 * 768 * 256;
    const int n_aout = 16 * 256 * 256;
    const int n_fus  = 4 * 256 * 1024;
    const int n_f1   = 4 * 1024 * 256;
    const int n_f2   = 4 * 256 * 1024;
    _Float16* hw_ip   = hmem;
    _Float16* hw_ain  = hw_ip   + n_ip;
    _Float16* hw_aout = hw_ain  + n_ain;
    _Float16* hw_fus  = hw_aout + n_aout;
    _Float16* hw_f1   = hw_fus  + n_fus;
    _Float16* hw_f2   = hw_f1   + n_f1;

    wconv_kernel<<<256, 256, 0, stream>>>(ip_w,       hw_ip,   n_ip);
    wconv_kernel<<<1024, 256, 0, stream>>>(attn_in_w,  hw_ain,  n_ain);
    wconv_kernel<<<512, 256, 0, stream>>>(attn_out_w, hw_aout, n_aout);
    wconv_kernel<<<512, 256, 0, stream>>>(fus_w,      hw_fus,  n_fus);
    wconv_kernel<<<512, 256, 0, stream>>>(ffn_w1,     hw_f1,   n_f1);
    wconv_kernel<<<512, 256, 0, stream>>>(ffn_w2,     hw_f2,   n_f2);

    #define LAUNCH_GEMM(KERN, A, LDA, ADDKP, W, LDW, BIAS, RESP, LDR, CP, LDC, COFF, NN, KK) \
        do { const int blocks_ = (MROWS / 16) * ((NN) / 64) / 4;                              \
             KERN<<<blocks_, 128, 0, stream>>>(A, LDA, ADDKP, W, LDW, BIAS, RESP, LDR,        \
                                               CP, LDC, COFF, NN, KK); } while (0)

    const int scales[4] = {5, 10, 20, 50};

    // x = emb @ ip_w.T + ip_b + PE
    LAUNCH_GEMM((gemm_wmma_kernel<false, false, 2>), emb, 640, nullptr,
                hw_ip, 640, ip_b, nullptr, 0, r_x, 256, 0, 256, 640);

    for (int l = 0; l < 4; l++) {
        for (int si = 0; si < 4; si++) {
            const int ls = l * 4 + si;
            LAUNCH_GEMM((gemm_wmma_kernel<true, false, 0>), r_x, 256, scale_emb + si * 256,
                        hw_ain + (size_t)ls * 768 * 256, 256,
                        attn_in_b + (size_t)ls * 768, nullptr, 0,
                        r_qkv, 768, 0, 768, 256);
            attn_kernel<<<BATCH * NHEAD * (SEQ / 16), 32, 0, stream>>>(r_qkv, r_o, scales[si]);
            LAUNCH_GEMM((gemm_wmma_kernel<false, false, 0>), r_o, 256, nullptr,
                        hw_aout + (size_t)ls * 256 * 256, 256,
                        attn_out_b + (size_t)ls * 256, nullptr, 0,
                        r_cc, 1024, si * 256, 256, 256);
        }
        LAUNCH_GEMM((gemm_wmma_kernel<false, true, 0>), r_cc, 1024, nullptr,
                    hw_fus + (size_t)l * 256 * 1024, 1024,
                    fus_b + l * 256, r_x, 256, r_tmp, 256, 0, 256, 1024);
        ln_kernel<<<MROWS / 8, 256, 0, stream>>>(r_tmp, r_x, n1_g + l * 256, n1_b + l * 256);
        LAUNCH_GEMM((gemm_wmma_kernel<false, false, 1>), r_x, 256, nullptr,
                    hw_f1 + (size_t)l * 1024 * 256, 256,
                    ffn_b1 + l * 1024, nullptr, 0, r_cc, 1024, 0, 1024, 256);
        LAUNCH_GEMM((gemm_wmma_kernel<false, true, 0>), r_cc, 1024, nullptr,
                    hw_f2 + (size_t)l * 256 * 1024, 1024,
                    ffn_b2 + l * 256, r_x, 256, r_tmp, 256, 0, 256, 1024);
        ln_kernel<<<MROWS / 8, 256, 0, stream>>>(r_tmp, r_x, n2_g + l * 256, n2_b + l * 256);
    }

    head_kernel<<<1, 256, 0, stream>>>(r_x, features,
        feat_w, feat_b, feat_ln_g, feat_ln_b,
        h1_w, h1_b, h_ln_g, h_ln_b, h2_w, h2_b, h3_w, h3_b,
        (float*)d_out);
}
